// SelfAttention3D_31696858644762
// MI455X (gfx1250) — matrix operand order repaired and checked by reading, not compiled, not run on hardware
//
#include <hip/hip_runtime.h>

// ---------------- constants ----------------
#define BWIN  1024
#define NTOK  128
#define DIM_  180
#define H_    6
#define HD    30
#define HDP   32      // head dim padded to 32 (WMMA K)
#define KQKV  192     // DIM_ padded to multiple of 32
#define NQKV  540
#define NQKVP 544     // padded to multiple of 16
#define DP    192     // DIM_ padded (FC K and activation width)
#define NW_   64
#define QK_SCALE 0.18257418583505536f  // 30^-0.5
#define QKV_STRIDE ((long)BWIN * H_ * NTOK * HDP)  // elements per q/k/v plane

typedef __attribute__((ext_vector_type(16))) __bf16 v16bf;
typedef __attribute__((ext_vector_type(8)))  __bf16 v8bf;
typedef __attribute__((ext_vector_type(8)))  float  v8f;

__device__ __forceinline__ __bf16 tobf(float f) { return (__bf16)f; }

__device__ __forceinline__ v8f vzero8() {
  v8f r = {0.f, 0.f, 0.f, 0.f, 0.f, 0.f, 0.f, 0.f};
  return r;
}

// A-matrix 16x32 bf16 fragment: per lane row = lane%16,
// K = {klo..klo+7} at p[0..7] and {klo+16..klo+23} at p[16..23].
__device__ __forceinline__ v16bf frag_a(const __bf16* p) {
  v8bf lo = *(const v8bf*)(p);
  v8bf hi = *(const v8bf*)(p + 16);
  v16bf r;
#pragma unroll
  for (int i = 0; i < 8; ++i) { r[i] = lo[i]; r[i + 8] = hi[i]; }
  return r;
}

// B-matrix 32x16 bf16 fragment: per lane col = lane%16,
// K = {klo..klo+7} at p[0..7] and {klo+16..klo+23} at p[16..23].
__device__ __forceinline__ v16bf frag_b(const __bf16* p) {
  v8bf lo = *(const v8bf*)(p);
  v8bf hi = *(const v8bf*)(p + 16);
  v16bf r;
#pragma unroll
  for (int i = 0; i < 8; ++i) { r[i] = lo[i]; r[i + 8] = hi[i]; }
  return r;
}

__device__ __forceinline__ v8f wmma_bf16(v16bf a, v16bf b, v8f c) {
  return __builtin_amdgcn_wmma_f32_16x16x32_bf16(false, a, false, b, (short)0, c,
                                                 false, false);
}

// ---- gfx1250 async global->LDS copy (16 bytes per lane, ASYNCcnt) ----
__device__ __forceinline__ void async_copy16(__bf16* lds_dst, const __bf16* gsrc) {
  unsigned ldsb = (unsigned)(unsigned long long)(uintptr_t)lds_dst;  // LDS byte offset
  asm volatile("global_load_async_to_lds_b128 %0, %1, off"
               :: "v"(ldsb), "v"(gsrc)
               : "memory");
}
__device__ __forceinline__ void async_wait0() {
  asm volatile("s_wait_asynccnt 0x0" ::: "memory");
}

// ---------------- kernel 1: weight transpose + bf16 + pad ----------------
__global__ void prep_kernel(const float* __restrict__ wqkv,
                            const float* __restrict__ wfc,
                            __bf16* __restrict__ wqkvT,
                            __bf16* __restrict__ wfcT) {
  const int total1 = NQKVP * KQKV;
  const int total2 = DP * DP;
  for (int i = blockIdx.x * blockDim.x + threadIdx.x; i < total1 + total2;
       i += gridDim.x * blockDim.x) {
    if (i < total1) {
      int c = i / KQKV, kk = i - c * KQKV;
      float val = (c < NQKV && kk < DIM_) ? wqkv[kk * NQKV + c] : 0.f;
      wqkvT[i] = tobf(val);
    } else {
      int j = i - total1;
      int c = j / DP, kk = j - c * DP;
      float val = (c < DIM_ && kk < DIM_) ? wfc[kk * DIM_ + c] : 0.f;
      wfcT[j] = tobf(val);
    }
  }
}

// ---------------- kernel 2: QKV projection (one window per block) --------
// qkv = single contiguous buffer: plane s (0=q,1=k,2=v) at s*QKV_STRIDE.
__global__ __launch_bounds__(256) void qkv_kernel(
    const float* __restrict__ x, const float* __restrict__ bqkv,
    const __bf16* __restrict__ wT, __bf16* __restrict__ qkv) {
  __shared__ __bf16 xs[NTOK * 200];  // [token][K=192], stride 200 (conflict-free)
  const int win = blockIdx.x;
  const int tid = threadIdx.x;

  for (int i = tid; i < NTOK * KQKV; i += 256) {
    int row = i / KQKV, kk = i - row * KQKV;
    float val = (kk < DIM_) ? x[((long)(win * NTOK + row)) * DIM_ + kk] : 0.f;
    xs[row * 200 + kk] = tobf(val);
  }
  __syncthreads();

  const int lane = tid & 31, wave = tid >> 5;
  const int mbase = wave * 16;
  const int arow = mbase + (lane & 15);
  const int kloA = (lane >> 4) * 8;
  const int kloB = (lane >> 4) * 8;
  const int nlane = lane & 15;

  // A fragments for this wave's row tile stay in registers for all columns.
  v16bf a[KQKV / 32];
#pragma unroll
  for (int kt = 0; kt < KQKV / 32; ++kt)
    a[kt] = frag_a(&xs[arow * 200 + kt * 32 + kloA]);

  auto loadB = [&](int ct, v16bf bb[KQKV / 32]) {
    const __bf16* bp = &wT[(long)(ct * 16 + nlane) * KQKV + kloB];
#pragma unroll
    for (int kt = 0; kt < KQKV / 32; ++kt) bb[kt] = frag_b(bp + kt * 32);
  };
  auto computeStore = [&](int ct, v16bf bb[KQKV / 32]) {
    v8f acc = vzero8();
#pragma unroll
    for (int kt = 0; kt < KQKV / 32; ++kt) acc = wmma_bf16(a[kt], bb[kt], acc);
    const int c = ct * 16 + nlane;
    if (c < NQKV) {
      const int s = c / DIM_;
      const int rem = c - s * DIM_;
      const int head = rem / HD;
      const int d = rem - head * HD;
      const float bias = bqkv[c];
      // Pure address arithmetic: no pointer select (avoids scratch spills).
      __bf16* dst = qkv + (long)s * QKV_STRIDE +
                    ((long)(win * H_ + head)) * NTOK * HDP;
#pragma unroll
      for (int r = 0; r < 8; ++r) {
        const int row = mbase + (lane >> 4) * 8 + r;
        float valf = acc[r] + bias;
        if (s == 0) valf *= QK_SCALE;
        dst[row * HDP + d] = tobf(valf);
        if (d == HD - 1) {  // zero the padded head-dim lanes
          dst[row * HDP + 30] = tobf(0.f);
          dst[row * HDP + 31] = tobf(0.f);
        }
      }
    }
  };

  // Software-pipelined over 34 column tiles: next tile's 12 B loads are
  // in flight while the current tile's 6 WMMAs execute.
  v16bf b0[KQKV / 32], b1[KQKV / 32];
  loadB(0, b0);
  for (int ct = 0; ct < NQKVP / 16; ct += 2) {
    loadB(ct + 1, b1);               // ct+1 <= 33: always valid
    computeStore(ct, b0);
    if (ct + 2 < NQKVP / 16) loadB(ct + 2, b0);
    computeStore(ct + 1, b1);
  }
}

// ---------------- kernel 3: attention (one window-head per block) --------
__global__ __launch_bounds__(256) void attn_kernel(
    const __bf16* __restrict__ q, const __bf16* __restrict__ k,
    const __bf16* __restrict__ v, const float* __restrict__ rpt,
    const float* __restrict__ mask, const int* __restrict__ rpi,
    __bf16* __restrict__ ao) {
  __shared__ __bf16 qs[NTOK * 40];   // [token][32] stride 40
  __shared__ __bf16 ks[NTOK * 40];
  __shared__ __bf16 vt[HDP * 136];   // V transposed: [d][token] stride 136
  __shared__ __bf16 ps[NTOK * 136];  // probabilities [row][col] stride 136

  const int win = blockIdx.x, h = blockIdx.y;
  const int tid = threadIdx.x;
  const long base = ((long)(win * H_ + h)) * NTOK * HDP;

  // Q/K tiles via async global->LDS DMA: 512 16-byte chunks each.
  for (int i = tid; i < 1024; i += 256) {
    const int which = i >> 9;      // 0 = q, 1 = k
    const int j = i & 511;
    const int row = j >> 2, c16 = j & 3;
    const __bf16* src = (which ? k : q) + base + row * HDP + c16 * 8;
    __bf16* dst = (which ? ks : qs) + row * 40 + c16 * 8;
    async_copy16(dst, src);
  }
  // V staged transposed (needs element shuffle -> VGPR path).
  for (int i = tid; i < NTOK * HDP; i += 256) {
    int row = i >> 5, d = i & 31;
    vt[d * 136 + row] = v[base + i];
  }
  async_wait0();
  __syncthreads();

  const int lane = tid & 31, wave = tid >> 5;
  const int nlane = lane & 15;
  const int mlocal0 = (lane >> 4) * 8;
  const int kloA = (lane >> 4) * 8;
  const int kloB = (lane >> 4) * 8;

  // ---- S = (scaled Q) x K^T : wave handles rows [wave*16, wave*16+16) ----
  v16bf aq = frag_a(&qs[(wave * 16 + nlane) * 40 + kloA]);
  v16bf bk[8];
#pragma unroll
  for (int nt = 0; nt < 8; ++nt)
    bk[nt] = frag_b(&ks[(nt * 16 + nlane) * 40 + kloB]);
  v8f acc[8];
#pragma unroll
  for (int nt = 0; nt < 8; ++nt) acc[nt] = wmma_bf16(aq, bk[nt], vzero8());

  // ---- add relative-position bias + window mask (register C-layout) ----
  const float* maskw = mask + (long)(win & (NW_ - 1)) * NTOK * NTOK;
#pragma unroll
  for (int nt = 0; nt < 8; ++nt) {
    const int col = nt * 16 + nlane;
#pragma unroll
    for (int r = 0; r < 8; ++r) {
      const int row = wave * 16 + mlocal0 + r;
      acc[nt][r] += rpt[rpi[row * NTOK + col] * H_ + h] + maskw[row * NTOK + col];
    }
  }

  // ---- register softmax: each row lives in one 16-lane half ----
#pragma unroll
  for (int r = 0; r < 8; ++r) {
    float m = acc[0][r];
#pragma unroll
    for (int nt = 1; nt < 8; ++nt) m = fmaxf(m, acc[nt][r]);
    m = fmaxf(m, __shfl_xor(m, 1, 16));
    m = fmaxf(m, __shfl_xor(m, 2, 16));
    m = fmaxf(m, __shfl_xor(m, 4, 16));
    m = fmaxf(m, __shfl_xor(m, 8, 16));
    float s = 0.f;
#pragma unroll
    for (int nt = 0; nt < 8; ++nt) {
      float e = __expf(acc[nt][r] - m);
      acc[nt][r] = e;
      s += e;
    }
    s += __shfl_xor(s, 1, 16);
    s += __shfl_xor(s, 2, 16);
    s += __shfl_xor(s, 4, 16);
    s += __shfl_xor(s, 8, 16);
    const float inv = 1.f / s;
    const int row = wave * 16 + mlocal0 + r;
#pragma unroll
    for (int nt = 0; nt < 8; ++nt)
      ps[row * 136 + nt * 16 + nlane] = tobf(acc[nt][r] * inv);
  }
  __syncthreads();

  // ---- O = P x V : M=128, N=32 (2 tiles), K=128 (4 steps) ----
  const int mt = wave;
  v16bf ap[4];  // P fragments shared by both output column tiles
#pragma unroll
  for (int kt = 0; kt < 4; ++kt)
    ap[kt] = frag_a(&ps[(mt * 16 + nlane) * 136 + kt * 32 + kloA]);
#pragma unroll
  for (int nt2 = 0; nt2 < 2; ++nt2) {
    v16bf bv[4];
#pragma unroll
    for (int kt = 0; kt < 4; ++kt)
      bv[kt] = frag_b(&vt[(nt2 * 16 + nlane) * 136 + kt * 32 + kloB]);
    v8f o = vzero8();
#pragma unroll
    for (int kt = 0; kt < 4; ++kt) o = wmma_bf16(ap[kt], bv[kt], o);

    const int d = nt2 * 16 + nlane;
    if (d < HD) {
      const int col = h * HD + d;
#pragma unroll
      for (int r = 0; r < 8; ++r) {
        const int row = mt * 16 + mlocal0 + r;
        ao[((long)(win * NTOK + row)) * DP + col] = tobf(o[r]);
      }
    }
  }

  // head-0 blocks zero the 12 padded activation columns for this window
  if (h == 0) {
    for (int i = tid; i < NTOK * (DP - DIM_); i += 256) {
      int row = i / (DP - DIM_), cc = DIM_ + i % (DP - DIM_);
      ao[((long)(win * NTOK + row)) * DP + cc] = tobf(0.f);
    }
  }
}

// ---------------- kernel 4: output FC (one window per block) -------------
__global__ __launch_bounds__(256) void fc_kernel(
    const __bf16* __restrict__ ain, const __bf16* __restrict__ wT,
    const float* __restrict__ bfc, float* __restrict__ out) {
  __shared__ __bf16 as[NTOK * 200];  // [token][K=192] stride 200
  const int win = blockIdx.x, tid = threadIdx.x;

  // Activation panel staged via async global->LDS: 128 rows x 24 chunks.
  for (int i = tid; i < NTOK * 24; i += 256) {
    const int row = i / 24, c16 = i % 24;
    const __bf16* src = ain + ((long)(win * NTOK + row)) * DP + c16 * 8;
    __bf16* dst = as + row * 200 + c16 * 8;
    async_copy16(dst, src);
  }
  async_wait0();
  __syncthreads();

  const int lane = tid & 31, wave = tid >> 5;
  const int mbase = wave * 16;
  const int arow = mbase + (lane & 15);
  const int kloA = (lane >> 4) * 8;
  const int kloB = (lane >> 4) * 8;
  const int nlane = lane & 15;

  v16bf a[DP / 32];
#pragma unroll
  for (int kt = 0; kt < DP / 32; ++kt)
    a[kt] = frag_a(&as[arow * 200 + kt * 32 + kloA]);

  auto loadB = [&](int ct, v16bf bb[DP / 32]) {
    const __bf16* bp = &wT[(long)(ct * 16 + nlane) * DP + kloB];
#pragma unroll
    for (int kt = 0; kt < DP / 32; ++kt) bb[kt] = frag_b(bp + kt * 32);
  };
  auto computeStore = [&](int ct, v16bf bb[DP / 32]) {
    v8f acc = vzero8();
#pragma unroll
    for (int kt = 0; kt < DP / 32; ++kt) acc = wmma_bf16(a[kt], bb[kt], acc);
    const int c = ct * 16 + nlane;
    if (c < DIM_) {
      const float bias = bfc[c];
#pragma unroll
      for (int r = 0; r < 8; ++r) {
        const int row = mbase + (lane >> 4) * 8 + r;
        out[((long)(win * NTOK + row)) * DIM_ + c] = acc[r] + bias;
      }
    }
  };

  // Software-pipelined over 12 column tiles.
  v16bf b0[DP / 32], b1[DP / 32];
  loadB(0, b0);
  for (int ct = 0; ct < DP / 16; ct += 2) {
    loadB(ct + 1, b1);               // ct+1 <= 11: always valid
    computeStore(ct, b0);
    if (ct + 2 < DP / 16) loadB(ct + 2, b0);
    computeStore(ct + 1, b1);
  }
}

// ---------------- host launcher ----------------
extern "C" void kernel_launch(void* const* d_in, const int* in_sizes, int n_in,
                              void* d_out, int out_size, void* d_ws, size_t ws_size,
                              hipStream_t stream) {
  const float* x     = (const float*)d_in[0];
  const float* w_qkv = (const float*)d_in[1];
  const float* b_qkv = (const float*)d_in[2];
  const float* w_fc  = (const float*)d_in[3];
  const float* b_fc  = (const float*)d_in[4];
  const float* rpt   = (const float*)d_in[5];
  const float* mask  = (const float*)d_in[6];
  const int*   rpi   = (const int*)d_in[7];
  float* out = (float*)d_out;

  char* ws = (char*)d_ws;
  const size_t WQKVT_OFF = 0;
  const size_t WQKVT_SZ  = (size_t)NQKVP * KQKV * 2;            // 208,896
  const size_t WFCT_OFF  = WQKVT_OFF + WQKVT_SZ;
  const size_t WFCT_SZ   = (size_t)DP * DP * 2;                 // 73,728
  const size_t QKV_SZ    = (size_t)QKV_STRIDE * 2;              // 50,331,648 each
  const size_t QKV_OFF   = WFCT_OFF + WFCT_SZ;                  // q,k,v contiguous
  const size_t AO_OFF    = QKV_OFF + 3 * QKV_SZ;

  __bf16* wqkvT = (__bf16*)(ws + WQKVT_OFF);
  __bf16* wfcT  = (__bf16*)(ws + WFCT_OFF);
  __bf16* qkvb  = (__bf16*)(ws + QKV_OFF);
  __bf16* qb    = qkvb;
  __bf16* kb    = qkvb + QKV_STRIDE;
  __bf16* vb    = qkvb + 2 * QKV_STRIDE;
  __bf16* ao    = (__bf16*)(ws + AO_OFF);

  prep_kernel<<<552, 256, 0, stream>>>(w_qkv, w_fc, wqkvT, wfcT);
  qkv_kernel<<<BWIN, 256, 0, stream>>>(x, b_qkv, wqkvT, qkvb);
  attn_kernel<<<dim3(BWIN, H_), 256, 0, stream>>>(qb, kb, vb, rpt, mask, rpi, ao);
  fc_kernel<<<BWIN, 256, 0, stream>>>(ao, wfcT, b_fc, out);
}
